// MultiheadAttention_33947421508336
// MI455X (gfx1250) — matrix operand order repaired and checked by reading, not compiled, not run on hardware
//
#include <hip/hip_runtime.h>

// ---------------------------------------------------------------------------
// Problem constants (from the reference)
// ---------------------------------------------------------------------------
constexpr int kB  = 2;
constexpr int kS  = 2048;
constexpr int kH  = 8;
constexpr int kDK = 64;
constexpr int kDV = 192;
constexpr int kC  = 1536;   // kH * kDV
constexpr int kKP = 512;    // kH * kDK
constexpr int kM  = kB * kS;     // 4096 total rows
constexpr int kRrows = 4096;     // 2S-1 = 4095, padded to 4096
constexpr int kMaxK = 1536;      // largest GEMM K dimension

typedef __attribute__((ext_vector_type(16))) __bf16 v16bf;
typedef __attribute__((ext_vector_type(8)))  float  v8f;
typedef unsigned short u16t;

// ---------------------------------------------------------------------------
// bf16 helpers / WMMA fragment loaders (layouts per CDNA5 ISA 7.12.2)
// ---------------------------------------------------------------------------
__device__ __forceinline__ u16t f2bf(float f) {
  unsigned u = __builtin_bit_cast(unsigned, f);
  u += 0x7FFFu + ((u >> 16) & 1u);          // round-to-nearest-even
  return (u16t)(u >> 16);
}
__device__ __forceinline__ float bf2f(u16t u) {
  unsigned v = ((unsigned)u) << 16;
  return __builtin_bit_cast(float, v);
}

struct alignas(16) U128 { unsigned w0, w1, w2, w3; };
union FragU { v16bf v; u16t us[16]; U128 q[2]; };

// A-matrix 16x32 bf16: lane holds row M=lane%16; elems 0..7 -> k = 8*half+i,
// elems 8..15 -> k = 16 + 8*half + i  (two contiguous 16B chunks)
__device__ __forceinline__ v16bf load_fragA(const u16t* rowp, int kb, int hf) {
  FragU f;
  f.q[0] = *(const U128*)(rowp + kb + 8 * hf);
  f.q[1] = *(const U128*)(rowp + kb + 16 + 8 * hf);
  return f.v;
}
// B-matrix 32x16 bf16: lane holds column N=lane%16; elems 0..7 -> k = 8*half+i,
// elems 8..15 -> k = 16 + 8*half + i  (two contiguous 16B chunks)
__device__ __forceinline__ v16bf load_fragB(const u16t* rowp, int kb, int hf) {
  FragU f;
  f.q[0] = *(const U128*)(rowp + kb + 8 * hf);
  f.q[1] = *(const U128*)(rowp + kb + 16 + 8 * hf);
  return f.v;
}

__device__ __forceinline__ v8f wmma_bf16(v16bf a, v16bf b, v8f c) {
  return __builtin_amdgcn_wmma_f32_16x16x32_bf16(
      /*neg_a=*/false, a, /*neg_b=*/false, b,
      /*c_mod=*/(short)0, c, /*reuse_a=*/false, /*reuse_b=*/false);
}

// ---------------------------------------------------------------------------
// Elementwise conversion kernels
// ---------------------------------------------------------------------------
__global__ void conv_f32_bf16(const float* __restrict__ in,
                              u16t* __restrict__ out, int n) {
  int i = blockIdx.x * blockDim.x + threadIdx.x;
  if (i < n) out[i] = f2bf(in[i]);
}

// in [K,N] fp32 row-major -> out [N,K] bf16 row-major
__global__ void transpose_f32_bf16(const float* __restrict__ in,
                                   u16t* __restrict__ out, int K, int N) {
  int i = blockIdx.x * blockDim.x + threadIdx.x;
  if (i >= K * N) return;
  int k = i / N, n = i - k * N;
  out[(size_t)n * K + k] = f2bf(in[i]);
}

// ---------------------------------------------------------------------------
// Positional features: r_k[r, c] = (pe @ Wr)[r, c], r in [0, 4095), padded row
// 4095 zeroed.  pe[f<96] = (cw[f] > |d|); pe[f>=96] = sign(d)*(cw[f-96] > |d|)
// with cw[j] = exp(ln(S+1)*(j+1)/96) - 1, d = r - (S-1).
// ---------------------------------------------------------------------------
__global__ void relpos_kernel(const float* __restrict__ Wr,
                              u16t* __restrict__ rkb) {
  int idx = blockIdx.x * blockDim.x + threadIdx.x;
  if (idx >= kRrows * kKP) return;
  int r = idx >> 9, c = idx & (kKP - 1);
  float val = 0.f;
  if (r < 2 * kS - 1) {
    float d   = (float)(r - (kS - 1));
    float ad  = fabsf(d);
    float sgn = (d > 0.f) ? 1.f : (d < 0.f ? -1.f : 0.f);
    float lp  = __logf((float)(kS + 1)) / 96.0f;
    for (int j = 0; j < 96; ++j) {
      float cw = __expf(lp * (float)(j + 1)) - 1.0f;
      if (cw > ad) val += Wr[j * kKP + c] + sgn * Wr[(96 + j) * kKP + c];
    }
  }
  rkb[idx] = f2bf(val);
}

// ---------------------------------------------------------------------------
// Generic bf16 WMMA GEMM:  C[M,N] = A[M,K] @ B[K,N], B given transposed (BT[N,K]).
// 4 waves/WG, each wave produces one 16x16 tile (block covers 16 x 64).
// The shared 16xK A-tile is staged into LDS once per WG with CDNA5 async
// copies (global_load_async_to_lds_b128, ASYNCcnt) -- data never touches VGPRs.
// mode 0: bf16 store          (k projection)
// mode 1: *DK^-0.5 + bias0/bias1 -> two bf16 outputs (q with r_w/r_r biases)
// mode 2: bf16 store transposed per head: out[b][h][dv][s]   (V)
// mode 3: fp32 store + bias0  (final output)
// ---------------------------------------------------------------------------
__global__ __launch_bounds__(128) void gemm_bf16_kernel(
    const u16t* __restrict__ A, const u16t* __restrict__ BT,
    int M, int N, int K, int mode,
    void* __restrict__ out0, void* __restrict__ out1,
    const float* __restrict__ bias0, const float* __restrict__ bias1) {
  __shared__ u16t aTile[16 * kMaxK];       // 48 KB

  int wave = threadIdx.x >> 5, lane = threadIdx.x & 31;
  int hf = lane >> 4, lr = lane & 15;
  int mt = blockIdx.x;
  int nt = blockIdx.y * 4 + wave;

  // ---- async-stage the 16 x K A tile into LDS (16B per lane per issue) ---
  {
    int chunksPerRow = K >> 3;             // 16B chunks per row
    int total = 16 * chunksPerRow;         // multiple of 128 for K in {512,1536}
    for (int c = threadIdx.x; c < total; c += 128) {
      int row = c / chunksPerRow, ch = c - row * chunksPerRow;
      const u16t* g = A + (size_t)(mt * 16 + row) * K + ch * 8;
      unsigned ldsOff = (unsigned)(size_t)&aTile[row * K + ch * 8];
      asm volatile("global_load_async_to_lds_b128 %0, %1, off"
                   :: "v"(ldsOff), "v"(g) : "memory");
    }
    asm volatile("s_wait_asynccnt 0x0" ::: "memory");
    __syncthreads();
  }

  const u16t* arow = &aTile[lr * K];
  const u16t* brow = BT + (size_t)(nt * 16 + lr) * K;
  v8f acc = {};
  for (int k = 0; k < K; k += 32)
    acc = wmma_bf16(load_fragA(arow, k, hf), load_fragB(brow, k, hf), acc);

  int col = nt * 16 + lr;
#pragma unroll
  for (int e = 0; e < 8; ++e) {
    int row  = mt * 16 + 8 * hf + e;       // C/D layout: M = e + 8*half, N = lane%16
    float vl = acc[e];
    if (mode == 0) {
      ((u16t*)out0)[(size_t)row * N + col] = f2bf(vl);
    } else if (mode == 1) {
      float q = vl * 0.125f;               // DK^-0.5, DK=64
      ((u16t*)out0)[(size_t)row * N + col] = f2bf(q + bias0[col]);
      ((u16t*)out1)[(size_t)row * N + col] = f2bf(q + bias1[col]);
    } else if (mode == 2) {
      int bb = row >> 11, ss = row & (kS - 1);
      int hh = col / kDV, dv = col - hh * kDV;
      ((u16t*)out0)[(((size_t)(bb * kH + hh)) * kDV + dv) * kS + ss] = f2bf(vl);
    } else {
      ((float*)out0)[(size_t)row * N + col] = vl + bias0[col];
    }
  }
}

// ---------------------------------------------------------------------------
// Fused attention: one WG = (b,h) x 16-query tile. 4 waves.
//   phase 1: each wave sweeps k-tiles (stride 4): content QK^T (2 WMMA) +
//            relative logits as two r-window WMMA tiles + diagonal gather
//            (relative_shift:  logits[q,k] += rel[q, k-q+S-1]); bf16 into LDS
//   phase 2: cooperative softmax; exp overwrites each bf16 slot in place
//   phase 3: P @ V with ds_load_b128 A fragments, 3 n-tiles per wave
// LDS ~73KB -> up to 4 concurrent WGs per WGP (320KB).
// ---------------------------------------------------------------------------
__global__ __launch_bounds__(128) void attn_kernel(
    const u16t* __restrict__ rq1, const u16t* __restrict__ rq2,
    const u16t* __restrict__ kb,  const u16t* __restrict__ vtb,
    const u16t* __restrict__ rkb, u16t* __restrict__ att) {
  __shared__ u16t  pl[16][kS];             // 64 KB: bf16 logits, then bf16 P
  __shared__ float relscr[4][16][32];      // per-wave rel gather scratch (8 KB)
  __shared__ float partials[16][8];
  __shared__ float rowmax[16];
  __shared__ float rowsum[16];

  int bh = blockIdx.x;
  int b = bh >> 3, h = bh & 7;
  int q0 = blockIdx.y * 16;
  int wave = threadIdx.x >> 5, lane = threadIdx.x & 31;
  int hf = lane >> 4, lr = lane & 15;

  // A fragments: rq1/rq2 rows for this query tile (persist in VGPRs)
  const u16t* r1row = rq1 + ((size_t)(b * kS + q0 + lr)) * kKP + h * kDK;
  const u16t* r2row = rq2 + ((size_t)(b * kS + q0 + lr)) * kKP + h * kDK;
  v16bf a1l = load_fragA(r1row, 0, hf), a1h = load_fragA(r1row, 32, hf);
  v16bf a2l = load_fragA(r2row, 0, hf), a2h = load_fragA(r2row, 32, hf);

  // ---- phase 1: logits tiles --------------------------------------------
  for (int kt = wave; kt < kS / 16; kt += 4) {
    int k0 = kt * 16;
    const u16t* krow = kb + ((size_t)(b * kS + k0 + lr)) * kKP + h * kDK;
    if (kt + 4 < kS / 16)  // prefetch next k-tile row for this wave
      __builtin_prefetch(kb + ((size_t)(b * kS + k0 + 64 + lr)) * kKP + h * kDK, 0, 1);

    v8f acc = {};
    acc = wmma_bf16(a1l, load_fragB(krow, 0, hf), acc);
    acc = wmma_bf16(a1h, load_fragB(krow, 32, hf), acc);

    // relative logits window: r in [base, base+32), base 16-aligned, >= 0
    int base = k0 - q0 + kS - 16;
    const u16t* w0row = rkb + ((size_t)(base + lr)) * kKP + h * kDK;
    const u16t* w1row = rkb + ((size_t)(base + 16 + lr)) * kKP + h * kDK;
    v8f rel0 = {}, rel1 = {};
    rel0 = wmma_bf16(a2l, load_fragB(w0row, 0, hf), rel0);
    rel0 = wmma_bf16(a2h, load_fragB(w0row, 32, hf), rel0);
    rel1 = wmma_bf16(a2l, load_fragB(w1row, 0, hf), rel1);
    rel1 = wmma_bf16(a2h, load_fragB(w1row, 32, hf), rel1);

    // scatter rel tiles to wave-private LDS, then diagonal gather
#pragma unroll
    for (int e = 0; e < 8; ++e) {
      relscr[wave][8 * hf + e][lr]      = rel0[e];
      relscr[wave][8 * hf + e][16 + lr] = rel1[e];
    }
    asm volatile("s_wait_dscnt 0" ::: "memory");   // wave-local LDS RAW fence
#pragma unroll
    for (int e = 0; e < 8; ++e) {
      int qi = 8 * hf + e;
      int j  = lr - qi + 15;                       // (k - q + S-1) - base, in [0,30]
      pl[qi][k0 + lr] = f2bf(acc[e] + relscr[wave][qi][j]);
    }
  }
  __syncthreads();

  // ---- phase 2: softmax over rows; exp overwrites bf16 slots in place ---
  {
    int row = threadIdx.x & 15, chunk = threadIdx.x >> 4;  // 8 chunks of 256
    float mx = -3.4e38f;
    for (int i = 0; i < 256; ++i) mx = fmaxf(mx, bf2f(pl[row][chunk * 256 + i]));
    partials[row][chunk] = mx;
    __syncthreads();
    if (chunk == 0) {
      float m = partials[row][0];
      for (int c2 = 1; c2 < 8; ++c2) m = fmaxf(m, partials[row][c2]);
      rowmax[row] = m;
    }
    __syncthreads();
    float rm = rowmax[row], s = 0.f;
    for (int i = 0; i < 256; ++i) {
      float e = __expf(bf2f(pl[row][chunk * 256 + i]) - rm);
      pl[row][chunk * 256 + i] = f2bf(e);          // same slot, same size: safe
      s += e;
    }
    partials[row][chunk] = s;
    __syncthreads();
    if (chunk == 0) {
      float t = 0.f;
      for (int c2 = 0; c2 < 8; ++c2) t += partials[row][c2];
      rowsum[row] = t;
    }
    __syncthreads();
  }

  // ---- phase 3: out = P @ V, 3 n-tiles of 16 per wave (N = 192) ---------
  const u16t* prow = &pl[lr][0];                   // ds_load_b128 fragments
  int n0 = wave * 48;
  const u16t* v0row = vtb + ((size_t)((b * kH + h) * kDV + n0 +  0 + lr)) * kS;
  const u16t* v1row = vtb + ((size_t)((b * kH + h) * kDV + n0 + 16 + lr)) * kS;
  const u16t* v2row = vtb + ((size_t)((b * kH + h) * kDV + n0 + 32 + lr)) * kS;
  v8f o0 = {}, o1 = {}, o2 = {};
  for (int ks = 0; ks < kS; ks += 32) {
    v16bf pf = load_fragA(prow, ks, hf);
    o0 = wmma_bf16(pf, load_fragB(v0row, ks, hf), o0);
    o1 = wmma_bf16(pf, load_fragB(v1row, ks, hf), o1);
    o2 = wmma_bf16(pf, load_fragB(v2row, ks, hf), o2);
  }
#pragma unroll
  for (int e = 0; e < 8; ++e) {
    int qi = 8 * hf + e;
    float inv = 1.0f / rowsum[qi];
    size_t base = ((size_t)(b * kS + q0 + qi)) * kC + h * kDV + n0 + lr;
    att[base +  0] = f2bf(o0[e] * inv);
    att[base + 16] = f2bf(o1[e] * inv);
    att[base + 32] = f2bf(o2[e] * inv);
  }
}

// ---------------------------------------------------------------------------
// Launcher
// ---------------------------------------------------------------------------
extern "C" void kernel_launch(void* const* d_in, const int* in_sizes, int n_in,
                              void* d_out, int out_size, void* d_ws, size_t ws_size,
                              hipStream_t stream) {
  (void)in_sizes; (void)n_in; (void)out_size; (void)ws_size;
  const float* x   = (const float*)d_in[0];
  const float* Wq  = (const float*)d_in[1];
  const float* Wk  = (const float*)d_in[2];
  const float* Wv  = (const float*)d_in[3];
  const float* Wr  = (const float*)d_in[4];
  const float* rwb = (const float*)d_in[5];   // [H*DK]
  const float* rrb = (const float*)d_in[6];   // [H*DK]
  const float* We  = (const float*)d_in[7];
  const float* be  = (const float*)d_in[8];   // [C]

  char* ws = (char*)d_ws;
  size_t off = 0;
  auto alloc = [&](size_t elems) {
    u16t* p = (u16t*)(ws + off);
    off = (off + elems * sizeof(u16t) + 255) & ~(size_t)255;
    return p;
  };
  u16t* xb   = alloc((size_t)kM * kC);
  u16t* WqT  = alloc((size_t)kKP * kC);
  u16t* WkT  = alloc((size_t)kKP * kC);
  u16t* WvT  = alloc((size_t)kC * kC);
  u16t* WeT  = alloc((size_t)kC * kC);
  u16t* rq1  = alloc((size_t)kM * kKP);
  u16t* rq2  = alloc((size_t)kM * kKP);
  u16t* kbuf = alloc((size_t)kM * kKP);
  u16t* vtb  = alloc((size_t)kM * kC);        // [B][H][DV][S]
  u16t* rkb  = alloc((size_t)kRrows * kKP);
  u16t* att  = alloc((size_t)kM * kC);

  dim3 blk128(128);
  // conversions
  conv_f32_bf16<<<(kM * kC) / 256, 256, 0, stream>>>(x, xb, kM * kC);
  transpose_f32_bf16<<<(kC * kKP) / 256, 256, 0, stream>>>(Wq, WqT, kC, kKP);
  transpose_f32_bf16<<<(kC * kKP) / 256, 256, 0, stream>>>(Wk, WkT, kC, kKP);
  transpose_f32_bf16<<<(kC * kC) / 256, 256, 0, stream>>>(Wv, WvT, kC, kC);
  transpose_f32_bf16<<<(kC * kC) / 256, 256, 0, stream>>>(We, WeT, kC, kC);
  relpos_kernel<<<(kRrows * kKP) / 256, 256, 0, stream>>>(Wr, rkb);

  // projections
  gemm_bf16_kernel<<<dim3(kM / 16, kKP / 64), blk128, 0, stream>>>(
      xb, WqT, kM, kKP, kC, 1, rq1, rq2, rwb, rrb);
  gemm_bf16_kernel<<<dim3(kM / 16, kKP / 64), blk128, 0, stream>>>(
      xb, WkT, kM, kKP, kC, 0, kbuf, nullptr, nullptr, nullptr);
  gemm_bf16_kernel<<<dim3(kM / 16, kC / 64), blk128, 0, stream>>>(
      xb, WvT, kM, kC, kC, 2, vtb, nullptr, nullptr, nullptr);

  // fused relative attention
  attn_kernel<<<dim3(kB * kH, kS / 16), blk128, 0, stream>>>(
      rq1, rq2, kbuf, vtb, rkb, att);

  // final projection -> fp32 output
  gemm_bf16_kernel<<<dim3(kM / 16, kC / 64), blk128, 0, stream>>>(
      att, WeT, kM, kC, kC, 3, d_out, nullptr, be, nullptr);
}
